// GlobalModel_60464549593523
// MI455X (gfx1250) — matrix operand order repaired and checked by reading, not compiled, not run on hardware
//
#include <hip/hip_runtime.h>
#include <hip/hip_bf16.h>

typedef __attribute__((ext_vector_type(16))) _Float16 v16h;
typedef __attribute__((ext_vector_type(8)))  _Float16 v8h;
typedef __attribute__((ext_vector_type(8)))  float    v8f;

#define LDH 128   // feature stride for node-major buffers

// ---- workspace layout ----
// f16 region (offsets in halfs): weights fragment-packed as WMMA B tiles
// [nt][kt][lane(32)][e(16)], zero-padded to Kt*32 in K.
// Conv weights use K reordered as [tap][ci] (k = t*CIN + ci).
#define OFF_C0   0        // conv L0: K=10 ->Kt=1, Nt=2  : 1024
#define OFF_C1   1024     // conv L1: K=160->Kt=5, Nt=2  : 5120
#define OFF_C2   6144
#define OFF_C3   11264
#define OFF_C4   16384
#define OFF_C5   21504    // conv L5: Kt=5, Nt=4 : 10240
#define OFF_G1WL 31744    // Kt=2, Nt=8 : 8192
#define OFF_G1WR 39936
#define OFF_G2WL 48128    // Kt=4, Nt=8 : 16384
#define OFF_G2WR 64512
#define OFF_LIN  80896    // [256]
// f16 region ends at 81152 halfs = 162,304 B

// f32 bias pool at byte offset 163840 (offsets in floats):
#define WSB_BYTE_OFF 163840
#define BP_CB0   0
#define BP_CB1   32
#define BP_CB2   64
#define BP_CB3   96
#define BP_CB4   128
#define BP_CB5   160      // 64 entries
#define BP_G1BL  224
#define BP_G1BR  352
#define BP_G1BI  480
#define BP_G2BL  608
#define BP_G2BR  736
#define BP_G2BI  864
#define BP_ATT1  992
#define BP_ATT2  1120
#define BP_TOTAL 1248     // floats (4992 B) -> ws total ~169 KB

struct SParams {
    const float* convW[6];
    const float* cb[6];
    const float* g1Wl; const float* g1Wr;
    const float* g2Wl; const float* g2Wr;
    const float* g1bl; const float* g1br; const float* g1att; const float* g1bias;
    const float* g2bl; const float* g2br; const float* g2att; const float* g2bias;
    const float* lin1w;
    _Float16*    wsh;
    float*       wsb;
};

struct KParams {
    const float* x;       // [N,1]
    const float* tstep;   // [B]
    const float* lin1b;   // [1]
    const _Float16* wsh;
    const float* wsb;
    float* out;           // [N,4]
};

// ---------------------------------------------------------------------------
// setup: f16 conversion + swizzle into WMMA B-fragment order + f32 bias pool.
// Fragment order: lane L, half e -> k = kt*32 + (L>>4)*16 + e, n = nt*16 + (L&15)
// (One-time kernel; static code size here is irrelevant to runtime.)
// ---------------------------------------------------------------------------
__device__ void pack_b(const float* __restrict__ src, _Float16* __restrict__ dst,
                       int K, int N, int tid, int nth)
{
    int Kt = (K + 31) >> 5;
    int total = (N >> 4) * Kt * 512;
    for (int j = tid; j < total; j += nth) {
        int nt = j / (Kt * 512);
        int r  = j - nt * (Kt * 512);
        int kt = r >> 9;
        int q  = r & 511;
        int L  = q >> 4, e = q & 15;
        int k  = kt * 32 + (L >> 4) * 16 + e;
        int n  = nt * 16 + (L & 15);
        dst[j] = (k < K) ? (_Float16)src[n * K + k] : (_Float16)0.f;
    }
}

// Conv: W[co][ci][t] row-major; GEMM K order is k = t*CIN + ci
__device__ void pack_conv(const float* __restrict__ src, _Float16* __restrict__ dst,
                          int CIN, int COUT, int tid, int nth)
{
    int K  = CIN * 5;
    int Kt = (K + 31) >> 5;
    int total = (COUT >> 4) * Kt * 512;
    for (int j = tid; j < total; j += nth) {
        int nt = j / (Kt * 512);
        int r  = j - nt * (Kt * 512);
        int kt = r >> 9;
        int q  = r & 511;
        int L  = q >> 4, e = q & 15;
        int k  = kt * 32 + (L >> 4) * 16 + e;
        int n  = nt * 16 + (L & 15);
        _Float16 v = (_Float16)0.f;
        if (k < K) {
            int t  = k / CIN;
            int ci = k - t * CIN;
            v = (_Float16)src[n * K + ci * 5 + t];
        }
        dst[j] = v;
    }
}

__device__ void copy_f(const float* __restrict__ src, float* __restrict__ dst,
                       int n, int tid, int nth)
{
    for (int j = tid; j < n; j += nth) dst[j] = src[j];
}

__global__ void convert_weights(SParams p)
{
    int tid = blockIdx.x * blockDim.x + threadIdx.x;
    int nth = gridDim.x * blockDim.x;
    pack_conv(p.convW[0], p.wsh + OFF_C0,  2, 32, tid, nth);
    pack_conv(p.convW[1], p.wsh + OFF_C1, 32, 32, tid, nth);
    pack_conv(p.convW[2], p.wsh + OFF_C2, 32, 32, tid, nth);
    pack_conv(p.convW[3], p.wsh + OFF_C3, 32, 32, tid, nth);
    pack_conv(p.convW[4], p.wsh + OFF_C4, 32, 32, tid, nth);
    pack_conv(p.convW[5], p.wsh + OFF_C5, 32, 64, tid, nth);
    pack_b(p.g1Wl, p.wsh + OFF_G1WL,  64, 128, tid, nth);
    pack_b(p.g1Wr, p.wsh + OFF_G1WR,  64, 128, tid, nth);
    pack_b(p.g2Wl, p.wsh + OFF_G2WL, 128, 128, tid, nth);
    pack_b(p.g2Wr, p.wsh + OFF_G2WR, 128, 128, tid, nth);
    for (int j = tid; j < 256; j += nth)
        p.wsh[OFF_LIN + j] = (_Float16)p.lin1w[j];
    // f32 bias/att pool
    for (int l = 0; l < 5; ++l) copy_f(p.cb[l], p.wsb + BP_CB0 + 32 * l, 32, tid, nth);
    copy_f(p.cb[5],   p.wsb + BP_CB5,  64, tid, nth);
    copy_f(p.g1bl,    p.wsb + BP_G1BL, 128, tid, nth);
    copy_f(p.g1br,    p.wsb + BP_G1BR, 128, tid, nth);
    copy_f(p.g1bias,  p.wsb + BP_G1BI, 128, tid, nth);
    copy_f(p.g2bl,    p.wsb + BP_G2BL, 128, tid, nth);
    copy_f(p.g2br,    p.wsb + BP_G2BR, 128, tid, nth);
    copy_f(p.g2bias,  p.wsb + BP_G2BI, 128, tid, nth);
    copy_f(p.g1att,   p.wsb + BP_ATT1, 128, tid, nth);
    copy_f(p.g2att,   p.wsb + BP_ATT2, 128, tid, nth);
}

// ---------------------------------------------------------------------------
// WMMA fragment loaders (16x16x32 f16, wave32; A and B both use contiguous k = 16*(lane>>4) + e)
// ---------------------------------------------------------------------------
__device__ __forceinline__ v16h combine8(v8h lo, v8h hi)
{
    v16h a;
#pragma unroll
    for (int e = 0; e < 8; ++e) { a[e] = lo[e]; a[8 + e] = hi[e]; }
    return a;
}

// A from node-major LDS [32][LD]: two 16-byte runs -> 2x ds_load_b128
template<int LD>
__device__ __forceinline__ v16h load_a_row(const _Float16* h, int row, int kt, int lane)
{
    const _Float16* rp = h + row * LD + kt * 32 + (lane >> 4) * 16;
    return combine8(*(const v8h*)(rp), *(const v8h*)(rp + 8));
}

// B from fragment-packed global: one contiguous 32-byte load per lane
__device__ __forceinline__ v16h load_b_frag(const _Float16* __restrict__ Wn,
                                            int kt, int lane)
{
    return *(const v16h*)(Wn + (kt * 32 + lane) * 16);
}

// Conv A. CIN==32: K order [tap][ci]; one K-tile == one tap; activations
// node-major [32][32] -> contiguous. CIN==2: act is the pre-built im2col
// buffer s_im0[32][32] (cols 0..9 filled, rest zero) -> also contiguous.
template<int CIN>
__device__ __forceinline__ v16h load_a_conv(const _Float16* act, int mt, int kt,
                                            int lane)
{
    int i = mt * 16 + (lane & 15);
    if constexpr (CIN == 32) {
        int row = (i + kt - 2) & 31;                  // circular pad by 2
        return load_a_row<32>(act, row, 0, lane);
    } else {
        return load_a_row<32>(act, i, 0, lane);       // im2col-ready
    }
}

// ---------------------------------------------------------------------------
// GEMM drivers (one 16x16 output tile per wave; B fragments hoisted to regs)
// ---------------------------------------------------------------------------
// Fused xl/xr projection sharing A fragments:
// [32 x K] @ [K x 128] twice (Wl, Wr) + bias -> f32 [32][128] each
template<int KT>
__device__ void gemm_h2(const _Float16* hb,
                        const _Float16* __restrict__ Wl,
                        const _Float16* __restrict__ Wr,
                        const float* bl, const float* br,
                        float* dl, float* dr, int tid)
{
    int lane = tid & 31, wave = tid >> 5;          // 8 waves; wave owns N-tile
    const _Float16* Wln = Wl + wave * (KT * 512);
    const _Float16* Wrn = Wr + wave * (KT * 512);
    v16h bfl[KT], bfr[KT];
#pragma unroll
    for (int kt = 0; kt < KT; ++kt) {
        bfl[kt] = load_b_frag(Wln, kt, lane);
        bfr[kt] = load_b_frag(Wrn, kt, lane);
    }
#pragma unroll
    for (int mt = 0; mt < 2; ++mt) {
        v8f accl = {}, accr = {};
#pragma unroll
        for (int kt = 0; kt < KT; ++kt) {
            v16h a = load_a_row<LDH>(hb, mt * 16 + (lane & 15), kt, lane);
            accl = __builtin_amdgcn_wmma_f32_16x16x32_f16(false, a, false, bfl[kt],
                                                          (short)0, accl, false, false);
            accr = __builtin_amdgcn_wmma_f32_16x16x32_f16(false, a, false, bfr[kt],
                                                          (short)0, accr, false, false);
        }
        int n   = wave * 16 + (lane & 15);
        float bnl = bl[n], bnr = br[n];            // LDS bias pool
        int mb  = mt * 16 + (lane >> 4) * 8;
#pragma unroll
        for (int r = 0; r < 8; ++r) {
            dl[(mb + r) * LDH + n] = accl[r] + bnl;
            dr[(mb + r) * LDH + n] = accr[r] + bnr;
        }
    }
}

// periodic conv layer as GEMM: [32 x CIN*5] @ [CIN*5 x COUT], bias + ELU,
// node-major output with stride LDOUT
template<int CIN, int COUT, int LDOUT>
__device__ void gemm_conv(const _Float16* act, const _Float16* __restrict__ Wt,
                          const float* bias, _Float16* actout, int tid)
{
    constexpr int K  = CIN * 5;
    constexpr int KT = (K + 31) / 32;
    constexpr int NT = COUT / 16;
    int lane = tid & 31, wave = tid >> 5;
    for (int t = wave; t < 2 * NT; t += 8) {
        int mt = t & 1, nt = t >> 1;
        const _Float16* Wn = Wt + nt * (KT * 512);
        v8f acc = {};
#pragma unroll
        for (int kt = 0; kt < KT; ++kt) {
            v16h a = load_a_conv<CIN>(act, mt, kt, lane);
            v16h b = load_b_frag(Wn, kt, lane);
            acc = __builtin_amdgcn_wmma_f32_16x16x32_f16(false, a, false, b,
                                                         (short)0, acc, false, false);
        }
        int n  = nt * 16 + (lane & 15);
        float bn = bias[n];
        int mb = mt * 16 + (lane >> 4) * 8;
#pragma unroll
        for (int r = 0; r < 8; ++r) {
            float v = acc[r] + bn;
            v = v > 0.f ? v : (__expf(v) - 1.0f);          // ELU
            actout[(mb + r) * LDOUT + n] = (_Float16)v;
        }
    }
}

__device__ __forceinline__ float gat_logit(const float* xls, const float* xr,
                                           const float* at)
{
    float s = 0.f;
#pragma unroll
    for (int c = 0; c < 32; ++c) {
        float v = xls[c] + xr[c];
        v = v > 0.f ? v : 0.2f * v;                        // leaky_relu(0.2)
        s += v * at[c];
    }
    return s;
}

// One GATv2 layer: projections via WMMA, 3-candidate softmax, aggregate, bias(+elu)
template<int KT, bool ELU>
__device__ void gat_layer(_Float16* s_h, const _Float16* Wl, const _Float16* Wr,
                          const float* blp, const float* brp, const float* attp,
                          const float* biasp,
                          float* s_xl, float* s_xo, const int* s_ind, int tid)
{
    gemm_h2<KT>(s_h, Wl, Wr, blp, brp, s_xl, s_xo, tid);  // xl, xr (shared A)
    __syncthreads();

    int i = 0, h = 0, src0 = 0, src1 = 0;
    float al0 = 0.f, al1 = 0.f, al2 = 0.f;
    if (tid < 128) {
        i = tid >> 2; h = tid & 3;
        int ind = s_ind[i];
        src0 = ind; src1 = (ind + 1) & 31;
        const float* xr = &s_xo[i * LDH + h * 32];
        const float* at = &attp[h * 32];
        float lg0 = -1e30f, lg1 = -1e30f;
        float lg2 = gat_logit(&s_xl[i * LDH + h * 32], xr, at);        // self-loop
        if (src0 != i) lg0 = gat_logit(&s_xl[src0 * LDH + h * 32], xr, at);
        if (src1 != i) lg1 = gat_logit(&s_xl[src1 * LDH + h * 32], xr, at);
        float m  = fmaxf(lg2, fmaxf(lg0, lg1));
        float e0 = __expf(lg0 - m), e1 = __expf(lg1 - m), e2 = __expf(lg2 - m);
        float inv = 1.0f / (e0 + e1 + e2);
        al0 = e0 * inv; al1 = e1 * inv; al2 = e2 * inv;
    }
    __syncthreads();                    // all xr reads done before out writes
    if (tid < 128) {
        float*       o  = &s_xo[i * LDH + h * 32];
        const float* x0 = &s_xl[src0 * LDH + h * 32];
        const float* x1 = &s_xl[src1 * LDH + h * 32];
        const float* x2 = &s_xl[i    * LDH + h * 32];
#pragma unroll
        for (int c = 0; c < 32; ++c)
            o[c] = al0 * x0[c] + al1 * x1[c] + al2 * x2[c];
    }
    __syncthreads();
    // bias (+ elu) -> next h (f16)
    for (int j = tid; j < 32 * LDH; j += 256) {
        int f = j & 127;
        float v = s_xo[j] + biasp[f];
        if (ELU) v = v > 0.f ? v : (__expf(v) - 1.0f);
        s_h[j] = (_Float16)v;
    }
    __syncthreads();
}

// ---------------------------------------------------------------------------
// fully fused model: one block per graph, all 4 steps, everything in LDS
// ---------------------------------------------------------------------------
__global__ __launch_bounds__(256)
void model_kernel(KParams p)
{
    __shared__ _Float16 s_h[32 * LDH];      // node-major features (f16)
    __shared__ float    s_xl[32 * LDH];     // xl projection (f32)
    __shared__ float    s_xo[32 * LDH];     // xr, then GAT output (f32)
    __shared__ _Float16 s_im0[32 * 32];     // layer-0 im2col buffer
    __shared__ _Float16 s_acta[32 * 32];    // conv ping (node-major [i][ci])
    __shared__ _Float16 s_actb[32 * 32];    // conv pong
    __shared__ float    s_bias[BP_TOTAL];   // bias/att pool (LDS copy)
    __shared__ float    s_x[32], s_xn[32];
    __shared__ int      s_ind[32];
    __shared__ float    s_d1[32], s_d2[32], s_ef[64], s_esum[32], s_deg[32];
    __shared__ _Float16 s_lin[256];

    const int b   = blockIdx.x;
    const int tid = threadIdx.x;
    const float ts = p.tstep[b];

    if (tid < 32) {
        int i = tid;
        s_x[i] = p.x[b * 32 + i];
        // searchsorted(X_GRID, wrap(xc - ts), 'right') - 1  (mod 32)
        float xn = (i + 0.5f) * (1.0f / 32.0f) - ts;
        xn -= floorf(xn);
        int t = (int)floorf(32.0f * xn - 0.5f);
        s_ind[i] = (t + 32) & 31;
    }
    s_lin[tid] = p.wsh[OFF_LIN + tid];
    for (int j = tid; j < BP_TOTAL; j += 256) s_bias[j] = p.wsb[j];
    for (int j = tid; j < 32 * 32; j += 256) s_im0[j] = (_Float16)0.f;
    __syncthreads();

    const float ch1 = -32.0f * ts;          // xi / STEP
    const float linb = p.lin1b[0];

    for (int step = 0; step < 4; ++step) {
        // layer-0 im2col: row i, K position k = t*2+ci (k<10), zeros elsewhere
        for (int j = tid; j < 320; j += 256) {
            int i = j / 10, k = j - 10 * i;
            int t = k >> 1;
            float v = (k & 1) ? ch1 : s_x[(i + t - 2) & 31];
            s_im0[i * 32 + k] = (_Float16)v;
        }
        __syncthreads();
        gemm_conv< 2, 32, 32>(s_im0,  p.wsh + OFF_C0, s_bias + BP_CB0, s_actb, tid); __syncthreads();
        gemm_conv<32, 32, 32>(s_actb, p.wsh + OFF_C1, s_bias + BP_CB1, s_acta, tid); __syncthreads();
        gemm_conv<32, 32, 32>(s_acta, p.wsh + OFF_C2, s_bias + BP_CB2, s_actb, tid); __syncthreads();
        gemm_conv<32, 32, 32>(s_actb, p.wsh + OFF_C3, s_bias + BP_CB3, s_acta, tid); __syncthreads();
        gemm_conv<32, 32, 32>(s_acta, p.wsh + OFF_C4, s_bias + BP_CB4, s_actb, tid); __syncthreads();
        // last conv writes node-major h directly (stride 128, feats 0..63)
        gemm_conv<32, 64, LDH>(s_actb, p.wsh + OFF_C5, s_bias + BP_CB5, s_h, tid); __syncthreads();

        gat_layer<2, false>(s_h, p.wsh + OFF_G1WL, p.wsh + OFF_G1WR,
                            s_bias + BP_G1BL, s_bias + BP_G1BR,
                            s_bias + BP_ATT1, s_bias + BP_G1BI,
                            s_xl, s_xo, s_ind, tid);
        gat_layer<4, true >(s_h, p.wsh + OFF_G2WL, p.wsh + OFF_G2WR,
                            s_bias + BP_G2BL, s_bias + BP_G2BR,
                            s_bias + BP_ATT2, s_bias + BP_G2BI,
                            s_xl, s_xo, s_ind, tid);

        // edge features: ef = h[src].w[0:128] + h[dst].w[128:256] + b
        if (tid < 64) {
            int i = tid >> 1, hh = tid & 1;
            const _Float16* w = &s_lin[hh * 128];
            float s = 0.f;
#pragma unroll 4
            for (int f = 0; f < 128; ++f)
                s += (float)s_h[i * LDH + f] * (float)w[f];
            float* dp = hh ? s_d2 : s_d1;
            dp[i] = s;
        }
        if (tid < 32) { s_esum[tid] = 0.f; s_deg[tid] = 0.f; }
        __syncthreads();

        if (tid < 64) {
            int i = tid >> 1, c = tid & 1;
            int src = (s_ind[i] + c) & 31;
            float ef = s_d1[src] + s_d2[i] + linb;
            s_ef[tid] = ef;
            atomicAdd(&s_esum[src], ef);       // segment_sum over src
            atomicAdd(&s_deg[src], 1.0f);      // degree over src
        }
        __syncthreads();

        if (tid < 32) {
            int i = tid;
            float acc = 0.f;
#pragma unroll
            for (int c = 0; c < 2; ++c) {
                int src = (s_ind[i] + c) & 31;
                float ef2 = s_ef[i * 2 + c] - s_esum[src] / s_deg[src]
                          + 1.0f / s_deg[src];
                acc += s_x[src] * ef2;
            }
            s_xn[i] = acc;
            p.out[(b * 32 + i) * 4 + step] = acc;
        }
        __syncthreads();
        if (tid < 32) s_x[tid] = s_xn[tid];
        __syncthreads();
    }
}

// ---------------------------------------------------------------------------
extern "C" void kernel_launch(void* const* d_in, const int* in_sizes, int n_in,
                              void* d_out, int out_size, void* d_ws, size_t ws_size,
                              hipStream_t stream)
{
    (void)n_in; (void)out_size; (void)ws_size;
    // d_in order: x, cur_time(unused), time_step, (W,b)x6, gat1{Wl,bl,Wr,br,att,bias},
    //             gat2{...}, lin1_w, lin1_b
    SParams sp;
    for (int l = 0; l < 6; ++l) {
        sp.convW[l] = (const float*)d_in[3 + 2 * l];
        sp.cb[l]    = (const float*)d_in[4 + 2 * l];
    }
    sp.g1Wl  = (const float*)d_in[15];
    sp.g1bl  = (const float*)d_in[16];
    sp.g1Wr  = (const float*)d_in[17];
    sp.g1br  = (const float*)d_in[18];
    sp.g1att = (const float*)d_in[19];
    sp.g1bias= (const float*)d_in[20];
    sp.g2Wl  = (const float*)d_in[21];
    sp.g2bl  = (const float*)d_in[22];
    sp.g2Wr  = (const float*)d_in[23];
    sp.g2br  = (const float*)d_in[24];
    sp.g2att = (const float*)d_in[25];
    sp.g2bias= (const float*)d_in[26];
    sp.lin1w = (const float*)d_in[27];
    sp.wsh = (_Float16*)d_ws;
    sp.wsb = (float*)((char*)d_ws + WSB_BYTE_OFF);
    convert_weights<<<96, 256, 0, stream>>>(sp);

    KParams kp;
    kp.x     = (const float*)d_in[0];
    kp.tstep = (const float*)d_in[2];
    kp.lin1b = (const float*)d_in[28];
    kp.wsh = (const _Float16*)d_ws;
    kp.wsb = (const float*)((char*)d_ws + WSB_BYTE_OFF);
    kp.out = (float*)d_out;

    int B = in_sizes[2];                     // number of graphs (1024)
    model_kernel<<<B, 256, 0, stream>>>(kp);
}
